// GNNEncoder_5677946765441
// MI455X (gfx1250) — matrix operand order repaired and checked by reading, not compiled, not run on hardware
//
#include <hip/hip_runtime.h>

#define CH 128  // channels

typedef __attribute__((ext_vector_type(16))) __bf16 v16bf;
typedef __attribute__((ext_vector_type(8)))  float  v8f;

// ---------------------------------------------------------------------------
// Fragment load: two b128 loads at base + compile-time byte offsets.
// A-fragments: 2nd quad at +32 bytes (K gap of 16 elements).
// B-fragments: 2nd quad at +32 bytes too (same K gap of 16 elements as A).
// ---------------------------------------------------------------------------
__device__ inline v16bf load_frag(const char* p, int off, int delta) {
    union { v16bf v; uint4 q[2]; } u;
    u.q[0] = *(const uint4*)(p + off);
    u.q[1] = *(const uint4*)(p + off + delta);
    return u.v;
}

// Generic loaders (tail path only).
__device__ inline v16bf load_a_frag(const __bf16* base, int ld, int m0, int k0, int lane) {
    const int half = lane >> 4;
    const int r    = lane & 15;
    const __bf16* p = base + (size_t)(m0 + r) * ld + k0 + half * 8;
    union { v16bf v; uint4 q[2]; } u;
    u.q[0] = *(const uint4*)(p);
    u.q[1] = *(const uint4*)(p + 16);
    return u.v;
}

__device__ inline v16bf load_b_frag(const __bf16* W, int ld, int n0, int k0, int lane) {
    const int half = lane >> 4;
    const int n    = lane & 15;
    const __bf16* p = W + (size_t)(n0 + n) * ld + k0 + half * 16;
    union { v16bf v; uint4 q[2]; } u;
    u.q[0] = *(const uint4*)(p);
    u.q[1] = *(const uint4*)(p + 8);
    return u.v;
}

// ---------------------------------------------------------------------------
// Small utility kernels
// ---------------------------------------------------------------------------

__global__ void zero_f32_kernel(float* p, int n) {
    int i = blockIdx.x * blockDim.x + threadIdx.x;
    if (i < n) p[i] = 0.0f;
}

__global__ void deg_kernel(const int* dst, float* deg, int E) {
    int e = blockIdx.x * blockDim.x + threadIdx.x;
    if (e < E) atomicAdd(&deg[dst[e]], 1.0f);
}

__global__ void conv_w_kernel(const float* Wl, const float* Wr,
                              __bf16* Wlb, __bf16* Wrb, int n) {
    int i = blockIdx.x * blockDim.x + threadIdx.x;
    if (i < n) {
        Wlb[i] = (__bf16)Wl[i];
        Wrb[i] = (__bf16)Wr[i];
    }
}

// One wave per edge: gather 128-float row of x[src], scatter-add into agg[dst].
// x and agg are L2-resident (51 MB each << 192 MB L2), so this stage is bound
// by L2 atomic throughput, not HBM.
__global__ void scatter_kernel(const float* __restrict__ x,
                               const int* __restrict__ src,
                               const int* __restrict__ dst,
                               float* __restrict__ agg, int E) {
    int wid  = (blockIdx.x * blockDim.x + threadIdx.x) >> 5;
    int lane = threadIdx.x & 31;
    if (wid >= E) return;
    int s = src[wid];
    int d = dst[wid];
    const float4 v = *(const float4*)(x + (size_t)s * CH + lane * 4);
    float* ap = agg + (size_t)d * CH + lane * 4;
    atomicAdd(ap + 0, v.x);
    atomicAdd(ap + 1, v.y);
    atomicAdd(ap + 2, v.z);
    atomicAdd(ap + 3, v.w);
}

// agg /= max(deg,1); produce bf16 copies of agg and x for the WMMA GEMM.
__global__ void finalize_kernel(const float* __restrict__ agg,
                                const float* __restrict__ deg,
                                const float* __restrict__ x,
                                __bf16* __restrict__ aggb,
                                __bf16* __restrict__ xb, int total) {
    int i = blockIdx.x * blockDim.x + threadIdx.x;
    if (i >= total) return;
    int row = i >> 7;  // /CH
    float dg = fmaxf(deg[row], 1.0f);
    aggb[i] = (__bf16)(agg[i] / dg);
    xb[i]   = (__bf16)x[i];
}

// ---------------------------------------------------------------------------
// Fused dual-GEMM + bias + BN-statistics kernel.
// Block = 256 threads = 8 waves; wave w owns output columns [16w, 16w+16).
// Block covers 128 rows x 128 cols; y = aggb@Wl^T + bl + xb@Wr^T (f32 accum).
//
// FULL=true: guard-free, fully unrolled, software-pipelined. All fragment
// addresses are (per-lane base pointer + compile-time immediate offset):
//   A frag (rt,kt): pA + rt*4096 + kt*64 (+32)   [row tile stride 16*256B]
//   B frag (kt):    pW + kt*64 (+32)
//   y store (rt,r): pY + rt*8192 + r*512
// so the inner stream is pure load-clause / wmma / store with no per-fragment
// VALU address chains.
// FULL=false: guarded dynamic loop for tail rows (1 block, perf-neutral).
// ---------------------------------------------------------------------------
template <bool FULL>
__global__ void __launch_bounds__(256)
gemm_bn_kernel(const __bf16* __restrict__ aggb,
               const __bf16* __restrict__ xb,
               const __bf16* __restrict__ Wlb,
               const __bf16* __restrict__ Wrb,
               const float* __restrict__ bl,
               float* __restrict__ y,
               float* __restrict__ bnsum,
               float* __restrict__ bnsumsq,
               int row_base, int nrows) {
    __shared__ float sred[2][CH];
    const int tid  = threadIdx.x;
    const int wave = tid >> 5;
    const int lane = tid & 31;
    const int half = lane >> 4;
    const int r    = lane & 15;
    const int n0   = wave * 16;
    const int col  = n0 + r;
    const int base = row_base + blockIdx.x * 128;

    // Per-lane base pointers (computed once).
    const char* pL = (const char*)Wlb + ((size_t)col * CH + half * 8) * 2;
    const char* pR = (const char*)Wrb + ((size_t)col * CH + half * 8) * 2;

    // Preload all B fragments (both weight matrices, 4 k-tiles each); reused
    // across the 8 row tiles of this block. Weights are L2-resident.
    v16bf bWl[4], bWr[4];
#pragma unroll
    for (int kt = 0; kt < 4; ++kt) {
        bWl[kt] = load_frag(pL, kt * 64, 32);
        bWr[kt] = load_frag(pR, kt * 64, 32);
    }

    const float bias = bl[col];
    float s = 0.0f, s2 = 0.0f;

    if (FULL) {
        const char* pA = (const char*)aggb + ((size_t)(base + r) * CH + half * 8) * 2;
        const char* pX = (const char*)xb   + ((size_t)(base + r) * CH + half * 8) * 2;
        float*      pY = y + (size_t)(base + half * 8) * CH + col;

        v16bf cA[4], cX[4], nA[4], nX[4];
#pragma unroll
        for (int kt = 0; kt < 4; ++kt) {
            cA[kt] = load_frag(pA, kt * 64, 32);
            cX[kt] = load_frag(pX, kt * 64, 32);
        }
#pragma unroll
        for (int rt = 0; rt < 8; ++rt) {
            if (rt < 7) {   // compile-time under full unroll
#pragma unroll
                for (int kt = 0; kt < 4; ++kt) {
                    nA[kt] = load_frag(pA, (rt + 1) * 4096 + kt * 64, 32);
                    nX[kt] = load_frag(pX, (rt + 1) * 4096 + kt * 64, 32);
                }
            }
            v8f acc0 = {};
            v8f acc1 = {};
#pragma unroll
            for (int kt = 0; kt < 4; ++kt) {
                acc0 = __builtin_amdgcn_wmma_f32_16x16x32_bf16(
                    false, cA[kt], false, bWl[kt], (short)0, acc0, false, false);
                acc1 = __builtin_amdgcn_wmma_f32_16x16x32_bf16(
                    false, cX[kt], false, bWr[kt], (short)0, acc1, false, false);
            }
#pragma unroll
            for (int rr = 0; rr < 8; ++rr) {
                float v = acc0[rr] + acc1[rr] + bias;
                pY[rt * 16 * CH + rr * CH] = v;     // imm offset rt*8192+rr*512
                s  += v;
                s2 += v * v;
            }
            if (rt < 7) {   // rotated away by unrolling (SSA rename)
#pragma unroll
                for (int kt = 0; kt < 4; ++kt) { cA[kt] = nA[kt]; cX[kt] = nX[kt]; }
            }
        }
    } else {
        // --- guarded tail path ---
        for (int rt = 0; rt < 8; ++rt) {
            const int m0 = base + rt * 16;           // uniform across block
            if (m0 >= nrows) break;                  // N % 16 == 0: tiles full
            v16bf aA[4], aX[4];
#pragma unroll
            for (int kt = 0; kt < 4; ++kt) {
                aA[kt] = load_a_frag(aggb, CH, m0, kt * 32, lane);
                aX[kt] = load_a_frag(xb,   CH, m0, kt * 32, lane);
            }
            v8f acc0 = {};
            v8f acc1 = {};
#pragma unroll
            for (int kt = 0; kt < 4; ++kt) {
                acc0 = __builtin_amdgcn_wmma_f32_16x16x32_bf16(
                    false, aA[kt], false, bWl[kt], (short)0, acc0, false, false);
                acc1 = __builtin_amdgcn_wmma_f32_16x16x32_bf16(
                    false, aX[kt], false, bWr[kt], (short)0, acc1, false, false);
            }
            const int row0 = m0 + half * 8;
#pragma unroll
            for (int rr = 0; rr < 8; ++rr) {
                float v = acc0[rr] + acc1[rr] + bias;
                y[(size_t)(row0 + rr) * CH + col] = v;
                s  += v;
                s2 += v * v;
            }
        }
    }

    // Block-level reduction of per-lane BN partial sums.
    if (tid < CH) { sred[0][tid] = 0.0f; sred[1][tid] = 0.0f; }
    __syncthreads();
    atomicAdd(&sred[0][col], s);     // ds_add_f32, 2-way conflict per column
    atomicAdd(&sred[1][col], s2);
    __syncthreads();
    if (tid < CH) {
        atomicAdd(&bnsum[tid],   sred[0][tid]);
        atomicAdd(&bnsumsq[tid], sred[1][tid]);
    }
}

// (y - mean) * rsqrt(var + eps) * gamma + beta, optional ReLU.
__global__ void bn_apply_kernel(const float* __restrict__ y,
                                const float* __restrict__ bnsum,
                                const float* __restrict__ bnsumsq,
                                const float* __restrict__ gamma,
                                const float* __restrict__ beta,
                                float* __restrict__ out,
                                int total, float invN, int relu) {
    int i = blockIdx.x * blockDim.x + threadIdx.x;
    if (i >= total) return;
    int c = i & (CH - 1);
    float mean = bnsum[c] * invN;
    float var  = bnsumsq[c] * invN - mean * mean;
    float v = (y[i] - mean) * rsqrtf(var + 1e-5f) * gamma[c] + beta[c];
    if (relu) v = fmaxf(v, 0.0f);
    out[i] = v;
}

// ---------------------------------------------------------------------------
// Host launcher
// ---------------------------------------------------------------------------
extern "C" void kernel_launch(void* const* d_in, const int* in_sizes, int n_in,
                              void* d_out, int out_size, void* d_ws, size_t ws_size,
                              hipStream_t stream) {
    const float* x_in  = (const float*)d_in[0];
    const int*   ei    = (const int*)d_in[1];
    const float* Wl    = (const float*)d_in[2];
    const float* bl    = (const float*)d_in[3];
    const float* Wr    = (const float*)d_in[4];
    const float* gamma = (const float*)d_in[5];
    const float* beta  = (const float*)d_in[6];

    const int N = in_sizes[0] / CH;           // 100000
    const int E = in_sizes[1] / 2;            // 600000
    const int L = in_sizes[2] / (CH * CH);    // 3
    const int* src = ei;
    const int* dst = ei + E;

    const size_t NC = (size_t)N * CH;

    // Carve scratch out of d_ws (256B aligned chunks).
    char* wp = (char*)d_ws;
    auto carve = [&](size_t bytes) {
        char* r = wp;
        wp += (bytes + 255) & ~(size_t)255;
        return (void*)r;
    };
    float*  xA      = (float*)carve(NC * 4);
    float*  xB      = (float*)carve(NC * 4);
    float*  agg     = (float*)carve(NC * 4);
    __bf16* aggb    = (__bf16*)carve(NC * 2);
    __bf16* xb      = (__bf16*)carve(NC * 2);
    __bf16* Wlb     = (__bf16*)carve((size_t)CH * CH * 2);
    __bf16* Wrb     = (__bf16*)carve((size_t)CH * CH * 2);
    float*  deg     = (float*)carve((size_t)N * 4);
    float*  bnsum   = (float*)carve(CH * 4);
    float*  bnsumsq = (float*)carve(CH * 4);

    const int TB = 256;
    const int totalNC = (int)NC;
    const int full_blocks = N / 128;          // guard-free blocks
    const int tail_base   = full_blocks * 128;
    const int has_tail    = (N > tail_base) ? 1 : 0;

    // Degree: dst is the same every layer, compute once.
    zero_f32_kernel<<<(N + TB - 1) / TB, TB, 0, stream>>>(deg, N);
    deg_kernel<<<(E + TB - 1) / TB, TB, 0, stream>>>(dst, deg, E);

    const float* x_cur = x_in;
    float* ybuf = xA;
    float* yalt = xB;

    for (int l = 0; l < L; ++l) {
        const int last = (l == L - 1);

        zero_f32_kernel<<<(totalNC + TB - 1) / TB, TB, 0, stream>>>(agg, totalNC);
        zero_f32_kernel<<<1, TB, 0, stream>>>(bnsum, CH);
        zero_f32_kernel<<<1, TB, 0, stream>>>(bnsumsq, CH);

        conv_w_kernel<<<(CH * CH + TB - 1) / TB, TB, 0, stream>>>(
            Wl + (size_t)l * CH * CH, Wr + (size_t)l * CH * CH, Wlb, Wrb, CH * CH);

        // One wave (32 lanes) per edge -> 8 edges per 256-thread block.
        scatter_kernel<<<(E + 7) / 8, TB, 0, stream>>>(x_cur, src, dst, agg, E);

        finalize_kernel<<<(totalNC + TB - 1) / TB, TB, 0, stream>>>(
            agg, deg, x_cur, aggb, xb, totalNC);

        // 128 rows x 128 cols per block; pipelined full blocks + guarded tail.
        if (full_blocks > 0) {
            gemm_bn_kernel<true><<<full_blocks, TB, 0, stream>>>(
                aggb, xb, Wlb, Wrb, bl + (size_t)l * CH, ybuf, bnsum, bnsumsq,
                0, N);
        }
        if (has_tail) {
            gemm_bn_kernel<false><<<1, TB, 0, stream>>>(
                aggb, xb, Wlb, Wrb, bl + (size_t)l * CH, ybuf, bnsum, bnsumsq,
                tail_base, N);
        }

        bn_apply_kernel<<<(totalNC + TB - 1) / TB, TB, 0, stream>>>(
            ybuf, bnsum, bnsumsq, gamma + (size_t)l * CH, beta + (size_t)l * CH,
            last ? (float*)d_out : ybuf, totalNC, 1.0f / (float)N, last ? 0 : 1);

        x_cur = ybuf;
        float* t = ybuf; ybuf = yalt; yalt = t;
    }
}
